// Attention_84696755077568
// MI455X (gfx1250) — matrix operand order repaired and checked by reading, not compiled, not run on hardware
//
#include <hip/hip_runtime.h>
#include <hip/hip_bf16.h>
#include <stdint.h>

typedef unsigned short u16;
typedef __attribute__((ext_vector_type(16))) __bf16 v16bf;
typedef __attribute__((ext_vector_type(8)))  float  v8f;

#define D_MODEL   1024
#define NUM_HEADS 16
#define D_KEY     64
#define SEQ       2048
#define BATCH     2
#define MROWS     (BATCH * SEQ)   // 4096

// ---------- helpers ----------

__device__ __forceinline__ u16 f2bf(float f) {
  union { float f; uint32_t u; } c; c.f = f;
  uint32_t u = c.u;
  uint32_t r = u + 0x7FFFu + ((u >> 16) & 1u);   // round-to-nearest-even
  return (u16)(r >> 16);
}

__device__ __forceinline__ v8f wmma_bf16(v16bf a, v16bf b, v8f c) {
  // 8 args: (neg_a, A, neg_b, B, c_mod, C, reuse_a, reuse_b)
  return __builtin_amdgcn_wmma_f32_16x16x32_bf16(false, a, false, b, (short)0, c,
                                                 false, false);
}

// B-fragment: 32 contiguous bytes (operand column K-contiguous), 16B aligned.
__device__ __forceinline__ v16bf load_frag_contig(const u16* p) {
  v16bf r;
  ((uint4*)&r)[0] = *(const uint4*)(p);
  ((uint4*)&r)[1] = *(const uint4*)(p + 8);
  return r;
}
// A-fragment: two 16B chunks: K = klow..klow+7 at p, K = klow+8..klow+15 at p+8 halves.
__device__ __forceinline__ v16bf load_frag_split(const u16* p) {
  v16bf r;
  ((uint4*)&r)[0] = *(const uint4*)(p);
  ((uint4*)&r)[1] = *(const uint4*)(p + 8);
  return r;
}

// ---------- f32 -> bf16 conversion ----------

__global__ void cvt_f32_bf16(const float* __restrict__ in, u16* __restrict__ out, int n) {
  int i = blockIdx.x * blockDim.x + threadIdx.x;
  int stride = gridDim.x * blockDim.x;
  for (; i < n; i += stride) out[i] = f2bf(in[i]);
}

// ---------- GEMM: C[M,N] = A[M,K](bf16) @ W[K,N](bf16) + bias(f32) ----------
// block = 256 threads = 8 waves; block tile 128(M) x 128(N); wave tile 32x64.
// 8 WMMAs per wave per 32-wide k-step.

#define WT_PITCH 40   // halves; 80B row stride: conflict-free reads, 16B aligned

template <bool BF16OUT>
__global__ __launch_bounds__(256)
void gemm_bf16_kernel(const u16* __restrict__ A, const u16* __restrict__ W,
                      const float* __restrict__ bias,
                      u16* __restrict__ Cb, float* __restrict__ Cf,
                      int M, int N, int K) {
  __shared__ u16 wt[128 * WT_PITCH];    // W tile transposed: wt[n][k]
  const int tid  = threadIdx.x;
  const int w    = tid >> 5, lane = tid & 31;
  const int half = lane >> 4, lm = lane & 15;
  const int wm = w & 3, wn = w >> 2;            // 4 m-waves x 2 n-waves
  const int m0 = blockIdx.y * 128 + wm * 32;
  const int n0blk = blockIdx.x * 128;
  const int n0 = n0blk + wn * 64;

  v8f acc[2][4];
  #pragma unroll
  for (int tm = 0; tm < 2; ++tm)
    #pragma unroll
    for (int tn = 0; tn < 4; ++tn) {
      float bv = bias[n0 + tn * 16 + lm];       // N col independent of element idx
      #pragma unroll
      for (int v = 0; v < 8; ++v) acc[tm][tn][v] = bv;
    }

  // co-op W-transpose mapping: k = lane, 16-wide n-strip per wave
  const int ldk = lane;                // 0..31
  const int ldn = w * 16;              // 0,16,...,112

  for (int k0 = 0; k0 < K; k0 += 32) {
    __syncthreads();
    { // load W[k0+ldk][n0blk+ldn .. +16] (32B) and scatter transposed into LDS
      const u16* src = W + (size_t)(k0 + ldk) * N + n0blk + ldn;
      uint4 d0 = *(const uint4*)src;
      uint4 d1 = *(const uint4*)(src + 8);
      uint32_t dv[8] = {d0.x, d0.y, d0.z, d0.w, d1.x, d1.y, d1.z, d1.w};
      #pragma unroll
      for (int j = 0; j < 8; ++j) {
        wt[(ldn + 2 * j + 0) * WT_PITCH + ldk] = (u16)(dv[j] & 0xFFFFu);
        wt[(ldn + 2 * j + 1) * WT_PITCH + ldk] = (u16)(dv[j] >> 16);
      }
    }
    __syncthreads();

    v16bf afrag[2], bfrag[4];
    #pragma unroll
    for (int tm = 0; tm < 2; ++tm)
      afrag[tm] = load_frag_split(A + (size_t)(m0 + tm * 16 + lm) * K + k0 + half * 16);
    #pragma unroll
    for (int tn = 0; tn < 4; ++tn)
      bfrag[tn] = load_frag_contig(&wt[(wn * 64 + tn * 16 + lm) * WT_PITCH + half * 16]);
    #pragma unroll
    for (int tm = 0; tm < 2; ++tm)
      #pragma unroll
      for (int tn = 0; tn < 4; ++tn)
        acc[tm][tn] = wmma_bf16(afrag[tm], bfrag[tn], acc[tm][tn]);
  }

  #pragma unroll
  for (int tm = 0; tm < 2; ++tm)
    #pragma unroll
    for (int tn = 0; tn < 4; ++tn) {
      const int col  = n0 + tn * 16 + lm;
      const int row0 = m0 + tm * 16 + half * 8;      // C layout
      size_t base = (size_t)row0 * N + col;
      if constexpr (BF16OUT) {
        u16* p = Cb + base;
        #pragma unroll
        for (int v = 0; v < 8; ++v) { *p = f2bf(acc[tm][tn][v]); p += N; }
      } else {
        float* p = Cf + base;
        #pragma unroll
        for (int v = 0; v < 8; ++v) { *p = acc[tm][tn][v]; p += N; }
      }
    }
}

// ---------- flash attention ----------
// block = 128 threads = 4 waves, all on one (b,h); each wave owns 16 q-rows.
// Key tiles of 32; V staged transposed in LDS; K read directly (K-contiguous rows).

#define VT_PITCH 40
#define PP       40

__global__ __launch_bounds__(128)
void attn_kernel(const u16* __restrict__ Qb, const u16* __restrict__ Kb,
                 const u16* __restrict__ Vb, u16* __restrict__ Ctx) {
  __shared__ u16 vt[64 * VT_PITCH];   // V tile transposed: vt[d][kk]
  __shared__ u16 ps[4 * 16 * PP];     // per-wave P scratch (C-layout -> A-layout)
  const int tid  = threadIdx.x;
  const int w    = tid >> 5, lane = tid & 31;
  const int half = lane >> 4, lm = lane & 15;
  const int bh = blockIdx.y;
  const int b = bh >> 4, h = bh & 15;
  const int q0 = blockIdx.x * 64 + w * 16;
  const size_t headoff = (size_t)b * SEQ * D_MODEL + (size_t)h * D_KEY;
  const u16* Qh = Qb + headoff;
  const u16* Kh = Kb + headoff;
  const u16* Vh = Vb + headoff;
  u16*       Oh = Ctx + headoff;

  // Q A-fragments for both 32-wide d-chunks (reused across all key tiles)
  v16bf qf[2];
  {
    const u16* qrow = Qh + (size_t)(q0 + lm) * D_MODEL;
    #pragma unroll
    for (int c = 0; c < 2; ++c)
      qf[c] = load_frag_split(qrow + c * 32 + half * 16);
  }

  v8f o[4];
  float mrow[8], lrow[8];
  #pragma unroll
  for (int v = 0; v < 8; ++v) { mrow[v] = -__builtin_inff(); lrow[v] = 0.f; }
  #pragma unroll
  for (int dt = 0; dt < 4; ++dt)
    #pragma unroll
    for (int v = 0; v < 8; ++v) o[dt][v] = 0.f;

  const float scale = 0.125f;         // 1/sqrt(64)
  u16* pw = &ps[w * 16 * PP];

  // co-op V-transpose mapping: key = lane, 16-wide d-strip per wave
  const int vkk = lane;               // 0..31
  const int vds = w * 16;             // 0,16,32,48

  for (int kt0 = 0; kt0 < SEQ; kt0 += 32) {
    __syncthreads();
    { // load V[kt0+vkk][h*64 + vds .. +16] (32B) and scatter transposed into LDS
      const u16* vsrc = Vh + (size_t)(kt0 + vkk) * D_MODEL + vds;
      uint4 d0 = *(const uint4*)vsrc;
      uint4 d1 = *(const uint4*)(vsrc + 8);
      uint32_t dv[8] = {d0.x, d0.y, d0.z, d0.w, d1.x, d1.y, d1.z, d1.w};
      #pragma unroll
      for (int j = 0; j < 8; ++j) {
        vt[(vds + 2 * j + 0) * VT_PITCH + vkk] = (u16)(dv[j] & 0xFFFFu);
        vt[(vds + 2 * j + 1) * VT_PITCH + vkk] = (u16)(dv[j] >> 16);
      }
    }
    __syncthreads();

    // S[16x32] = Q @ K^T  (two 16x16 accumulators, contraction over d=64)
    v8f s0, s1;
    #pragma unroll
    for (int v = 0; v < 8; ++v) { s0[v] = 0.f; s1[v] = 0.f; }
    #pragma unroll
    for (int c = 0; c < 2; ++c) {
      const u16* kp0 = Kh + (size_t)(kt0 +      lm) * D_MODEL + c * 32 + half * 16;
      const u16* kp1 = Kh + (size_t)(kt0 + 16 + lm) * D_MODEL + c * 32 + half * 16;
      s0 = wmma_bf16(qf[c], load_frag_contig(kp0), s0);
      s1 = wmma_bf16(qf[c], load_frag_contig(kp1), s1);
    }

    // online softmax (rows split across 16-lane halves -> xor-shuffle reduce)
    #pragma unroll
    for (int v = 0; v < 8; ++v) {
      float a0 = s0[v] * scale, a1 = s1[v] * scale;
      float t = fmaxf(a0, a1);
      t = fmaxf(t, __shfl_xor(t, 1));
      t = fmaxf(t, __shfl_xor(t, 2));
      t = fmaxf(t, __shfl_xor(t, 4));
      t = fmaxf(t, __shfl_xor(t, 8));
      float nm = fmaxf(mrow[v], t);
      float f  = __expf(mrow[v] - nm);
      mrow[v] = nm;
      float p0 = __expf(a0 - nm);
      float p1 = __expf(a1 - nm);
      float sum = p0 + p1;
      sum += __shfl_xor(sum, 1);
      sum += __shfl_xor(sum, 2);
      sum += __shfl_xor(sum, 4);
      sum += __shfl_xor(sum, 8);
      lrow[v] = lrow[v] * f + sum;
      #pragma unroll
      for (int dt = 0; dt < 4; ++dt) o[dt][v] *= f;
      int r = half * 8 + v;                          // C-layout row
      pw[r * PP +      lm] = f2bf(p0);
      pw[r * PP + 16 + lm] = f2bf(p1);
    }
    __builtin_amdgcn_wave_barrier();
    asm volatile("s_wait_dscnt 0" ::: "memory");     // wave-local LDS RAW

    // P A-fragment from LDS scratch, then O += P @ V
    v16bf pf = load_frag_split(&pw[lm * PP + half * 16]);
    #pragma unroll
    for (int dt = 0; dt < 4; ++dt) {
      const u16* vp = &vt[(dt * 16 + lm) * VT_PITCH + half * 16];
      o[dt] = wmma_bf16(pf, load_frag_contig(vp), o[dt]);
    }
  }

  // epilogue: O /= l, write bf16 ctx (row-major [B*S, D_MODEL], head slice)
  #pragma unroll
  for (int v = 0; v < 8; ++v) {
    float rl = 1.0f / lrow[v];
    int r = half * 8 + v;
    u16* orow = Oh + (size_t)(q0 + r) * D_MODEL;
    #pragma unroll
    for (int dt = 0; dt < 4; ++dt)
      orow[dt * 16 + lm] = f2bf(o[dt][v] * rl);
  }
}

// ---------- launch ----------

extern "C" void kernel_launch(void* const* d_in, const int* in_sizes, int n_in,
                              void* d_out, int out_size, void* d_ws, size_t ws_size,
                              hipStream_t stream) {
  const float* x   = (const float*)d_in[0];
  const float* W_q = (const float*)d_in[1];
  const float* b_q = (const float*)d_in[2];
  const float* W_k = (const float*)d_in[3];
  const float* b_k = (const float*)d_in[4];
  const float* W_v = (const float*)d_in[5];
  const float* b_v = (const float*)d_in[6];
  const float* W_o = (const float*)d_in[7];
  const float* b_o = (const float*)d_in[8];
  float* out = (float*)d_out;

  char* ws = (char*)d_ws;                       // 48 MB used
  u16* xb  = (u16*)(ws + (size_t)0);            //  8 MB: x bf16   [4096,1024]
  u16* wqb = (u16*)(ws + ((size_t)8  << 20));   //  2 MB each
  u16* wkb = (u16*)(ws + ((size_t)10 << 20));
  u16* wvb = (u16*)(ws + ((size_t)12 << 20));
  u16* wob = (u16*)(ws + ((size_t)14 << 20));
  u16* Qb  = (u16*)(ws + ((size_t)16 << 20));   //  8 MB each
  u16* Kb  = (u16*)(ws + ((size_t)24 << 20));
  u16* Vb  = (u16*)(ws + ((size_t)32 << 20));
  u16* Cx  = (u16*)(ws + ((size_t)40 << 20));

  const int nx = MROWS * D_MODEL;       // 4,194,304
  const int nw = D_MODEL * D_MODEL;     // 1,048,576
  cvt_f32_bf16<<<1024, 256, 0, stream>>>(x,   xb,  nx);
  cvt_f32_bf16<<<512,  256, 0, stream>>>(W_q, wqb, nw);
  cvt_f32_bf16<<<512,  256, 0, stream>>>(W_k, wkb, nw);
  cvt_f32_bf16<<<512,  256, 0, stream>>>(W_v, wvb, nw);
  cvt_f32_bf16<<<512,  256, 0, stream>>>(W_o, wob, nw);

  dim3 gg(D_MODEL / 128, MROWS / 128);  // (8, 32)
  gemm_bf16_kernel<true ><<<gg, 256, 0, stream>>>(xb, wqb, b_q, Qb, nullptr, MROWS, D_MODEL, D_MODEL);
  gemm_bf16_kernel<true ><<<gg, 256, 0, stream>>>(xb, wkb, b_k, Kb, nullptr, MROWS, D_MODEL, D_MODEL);
  gemm_bf16_kernel<true ><<<gg, 256, 0, stream>>>(xb, wvb, b_v, Vb, nullptr, MROWS, D_MODEL, D_MODEL);

  dim3 ga(SEQ / 64, BATCH * NUM_HEADS); // (32, 32)
  attn_kernel<<<ga, 128, 0, stream>>>(Qb, Kb, Vb, Cx);

  gemm_bf16_kernel<false><<<gg, 256, 0, stream>>>(Cx, wob, b_o, nullptr, out, MROWS, D_MODEL, D_MODEL);
}
